// GroupedQueryAttention_56564719288415
// MI455X (gfx1250) — matrix operand order repaired and checked by reading, not compiled, not run on hardware
//
#include <hip/hip_runtime.h>
#include <hip/hip_bf16.h>

// ---------------------------------------------------------------------------
// GQA attention layer for MI455X (gfx1250, wave32, WMMA + async LDS copies).
// B=2, T=2048, D=2048, H=16, HKV=4, HD=128, THETA=10000
// ---------------------------------------------------------------------------

#define B_ 2
#define T_ 2048
#define D_ 2048
#define H_ 16
#define HKV_ 4
#define HD_ 128

typedef __attribute__((ext_vector_type(16))) __bf16 bf16x16;
typedef __attribute__((ext_vector_type(8)))  float  f32x8;

static __device__ __forceinline__ __bf16 us2bf(unsigned short u) {
    union { unsigned short u; __bf16 b; } x; x.u = u; return x.b;
}
static __device__ __forceinline__ unsigned short f2us(float f) {
    union { unsigned short u; __bf16 b; } x; x.b = (__bf16)f; return x.u;
}
// Packed pair conversion -> single v_cvt_pk_bf16_f32
static __device__ __forceinline__ unsigned pk2(float a, float b) {
    union { __hip_bfloat162 h; unsigned u; } x;
    x.h = __float22bfloat162_rn(make_float2(a, b));
    return x.u;
}
// Unpack 8 packed bf16 (uint4) into fragment elements [base, base+8)
static __device__ __forceinline__ void unpack8(uint4 v, bf16x16 &f, int base) {
    unsigned w[4] = { v.x, v.y, v.z, v.w };
#pragma unroll
    for (int j = 0; j < 4; ++j) {
        f[base + 2*j]     = us2bf((unsigned short)(w[j] & 0xffffu));
        f[base + 2*j + 1] = us2bf((unsigned short)(w[j] >> 16));
    }
}
// Load a 16-element bf16 B/A fragment from LDS bf16 values p[0..8) and p[16..24).
static __device__ __forceinline__ void ldfrag(const unsigned short* p, bf16x16 &f) {
    unpack8(reinterpret_cast<const uint4*>(p)[0], f, 0);
    unpack8(reinterpret_cast<const uint4*>(p)[2], f, 8);
}
// Raw LDS byte offset of a __shared__ pointer (low 32 bits of flat address).
static __device__ __forceinline__ unsigned lds_off(const void* p) {
    return (unsigned)(unsigned long long)(uintptr_t)p;
}

// ---------------------------------------------------------------------------
// Generic GEMM: C[M,N] (f32) = A[M,K] (f32) x B[K,N] (f32), via bf16 WMMA.
// Block: 128 threads (4 waves). Tile: 64x64, K stepped by 32.
// Software-pipelined: double-buffered LDS, next-tile global loads overlap
// the WMMA compute phase; one barrier per k-step; B fragments rotate
// through two register buffers so DS waits are partial, not full drains.
// ---------------------------------------------------------------------------
__global__ __launch_bounds__(128)
void gemm_bf16_wmma(const float* __restrict__ A, const float* __restrict__ Bm,
                    float* __restrict__ C, int M, int N, int K) {
    __shared__ __align__(16) unsigned short Asm[2][64 * 32];  // [m][k]
    __shared__ __align__(16) unsigned short Bts[2][64 * 32];  // [n][k] transposed

    const int tid  = threadIdx.x;
    const int wave = tid >> 5;
    const int lane = tid & 31;
    const int mBase = blockIdx.y * 64;
    const int nBase = blockIdx.x * 64;

    f32x8 acc[4];
#pragma unroll
    for (int n = 0; n < 4; ++n)
#pragma unroll
        for (int r = 0; r < 8; ++r) acc[n][r] = 0.0f;

    const int ml   = (lane & 15) + wave * 16;   // A-fragment row
    const int koff = (lane < 16) ? 0 : 8;       // A-layout half-wave K offset
    const int kb2  = (lane >> 4) << 3;          // B-layout half-wave K offset (same as A)
    const int colN = lane & 15;

    // per-thread staging coordinates (invariant across k-steps)
    int am[4], ac4[4], bk[4], bc4[4];
#pragma unroll
    for (int it = 0; it < 4; ++it) {
        int g = tid + it * 128;
        am[it]  = g >> 3;
        ac4[it] = (g & 7) << 2;
        bk[it]  = g >> 4;
        bc4[it] = (g & 15) << 2;
    }

    // ---- prologue: load tile k0=0 into registers ----
    float4 fa[4], fb[4];
#pragma unroll
    for (int it = 0; it < 4; ++it)
        fa[it] = *reinterpret_cast<const float4*>(
            A + (size_t)(mBase + am[it]) * K + ac4[it]);
#pragma unroll
    for (int it = 0; it < 4; ++it)
        fb[it] = *reinterpret_cast<const float4*>(
            Bm + (size_t)bk[it] * N + nBase + bc4[it]);

    int buf = 0;
    for (int k0 = 0; k0 < K; k0 += 32) {
        // ---- store staged registers -> LDS[buf] (f32 -> bf16) ----
#pragma unroll
        for (int it = 0; it < 4; ++it) {
            uint2 p; p.x = pk2(fa[it].x, fa[it].y); p.y = pk2(fa[it].z, fa[it].w);
            *reinterpret_cast<uint2*>(&Asm[buf][am[it] * 32 + ac4[it]]) = p;
        }
#pragma unroll
        for (int it = 0; it < 4; ++it) {
            Bts[buf][(bc4[it] + 0) * 32 + bk[it]] = f2us(fb[it].x);
            Bts[buf][(bc4[it] + 1) * 32 + bk[it]] = f2us(fb[it].y);
            Bts[buf][(bc4[it] + 2) * 32 + bk[it]] = f2us(fb[it].z);
            Bts[buf][(bc4[it] + 3) * 32 + bk[it]] = f2us(fb[it].w);
        }
        __syncthreads();

        // ---- issue next-tile global loads (overlap with WMMA below) ----
        if (k0 + 32 < K) {
            const int kn = k0 + 32;
#pragma unroll
            for (int it = 0; it < 4; ++it)
                fa[it] = *reinterpret_cast<const float4*>(
                    A + (size_t)(mBase + am[it]) * K + kn + ac4[it]);
#pragma unroll
            for (int it = 0; it < 4; ++it)
                fb[it] = *reinterpret_cast<const float4*>(
                    Bm + (size_t)(kn + bk[it]) * N + nBase + bc4[it]);
        }

        // ---- A fragment (16x32) ----
        bf16x16 a;
        {
            const unsigned short* ap = &Asm[buf][ml * 32];
            uint4 v0 = *reinterpret_cast<const uint4*>(ap + koff);
            uint4 v1 = *reinterpret_cast<const uint4*>(ap + 16 + koff);
            unpack8(v0, a, 0);
            unpack8(v1, a, 8);
        }
        // ---- B fragments: 2-buffer rotation, WMMAs interleaved ----
        {
            const unsigned short* base = &Bts[buf][colN * 32 + kb2];
            bf16x16 b0, b1;
            ldfrag(base + 0 * 512, b0);
            ldfrag(base + 1 * 512, b1);
            acc[0] = __builtin_amdgcn_wmma_f32_16x16x32_bf16(
                false, a, false, b0, (short)0, acc[0], false, false);
            ldfrag(base + 2 * 512, b0);
            acc[1] = __builtin_amdgcn_wmma_f32_16x16x32_bf16(
                false, a, false, b1, (short)0, acc[1], false, false);
            ldfrag(base + 3 * 512, b1);
            acc[2] = __builtin_amdgcn_wmma_f32_16x16x32_bf16(
                false, a, false, b0, (short)0, acc[2], false, false);
            acc[3] = __builtin_amdgcn_wmma_f32_16x16x32_bf16(
                false, a, false, b1, (short)0, acc[3], false, false);
        }

        buf ^= 1;
    }

    // ---- store C (f32) ----
    const int rowh = (lane >> 4) << 3;
#pragma unroll
    for (int n = 0; n < 4; ++n)
#pragma unroll
        for (int r = 0; r < 8; ++r)
            C[(size_t)(mBase + wave * 16 + rowh + r) * N + nBase + n * 16 + colN]
                = acc[n][r];
}

// ---------------------------------------------------------------------------
// RoPE: read f32 [b*T+t][heads*HD], apply rotation, write bf16 [b,h,t,hd].
// One thread per (b, h, t, i) pair, i in [0,64).
// ---------------------------------------------------------------------------
__global__ __launch_bounds__(256)
void rope_to_bf16(const float* __restrict__ in, unsigned short* __restrict__ out,
                  int heads, int total) {
    int idx = blockIdx.x * blockDim.x + threadIdx.x;
    if (idx >= total) return;
    int i   = idx & 63;
    int tmp = idx >> 6;
    int t   = tmp & (T_ - 1);
    tmp >>= 11;
    int h = tmp % heads;
    int b = tmp / heads;

    // inv_freq = theta^(-2i/HD) = exp(-i * ln(theta)/64)
    float inv = __expf(-(float)i * (9.2103403719761836f / 64.0f));
    float ang = (float)t * inv;
    float s, c;
    __sincosf(ang, &s, &c);

    size_t ib = (size_t)(b * T_ + t) * (heads * HD_) + h * HD_;
    float x1 = in[ib + i];
    float x2 = in[ib + 64 + i];
    size_t ob = ((size_t)(b * heads + h) * T_ + t) * HD_;
    out[ob + i]      = f2us(x1 * c - x2 * s);
    out[ob + 64 + i] = f2us(x2 * c + x1 * s);
}

// V layout: f32 [b*T+t][HKV*HD] -> bf16 [b,kvh,t,hd]
__global__ __launch_bounds__(256)
void v_to_bf16(const float* __restrict__ in, unsigned short* __restrict__ out,
               int total) {
    int idx = blockIdx.x * blockDim.x + threadIdx.x;
    if (idx >= total) return;
    int hd  = idx & (HD_ - 1);
    int tmp = idx >> 7;
    int t   = tmp & (T_ - 1);
    tmp >>= 11;
    int h = tmp % HKV_;
    int b = tmp / HKV_;
    out[((size_t)(b * HKV_ + h) * T_ + t) * HD_ + hd] =
        f2us(in[(size_t)(b * T_ + t) * (HKV_ * HD_) + h * HD_ + hd]);
}

// ---------------------------------------------------------------------------
// Flash attention (GQA, online softmax). Block = 128 threads = 4 waves.
// Each wave owns a 16-row Q tile; block covers 64 rows.
// K tile staged with GLOBAL_LOAD_ASYNC_TO_LDS_B128 (ASYNCcnt path);
// V tile staged transposed via regular loads + DS stores.
// ---------------------------------------------------------------------------
__global__ __launch_bounds__(128)
void gqa_flash_attn(const unsigned short* __restrict__ qb,
                    const unsigned short* __restrict__ kb,
                    const unsigned short* __restrict__ vb,
                    const int* __restrict__ amask,
                    const int* __restrict__ causal_p,
                    float* __restrict__ attnf) {
    __shared__ __align__(16) unsigned short Kt[32 * 128];   // [key][hd]
    __shared__ __align__(16) unsigned short Vt[128 * 32];   // [hd][key]
    __shared__ __align__(16) unsigned short Ps[4][16 * 32]; // per-wave P

    const int tid  = threadIdx.x;
    const int wave = tid >> 5;
    const int lane = tid & 31;
    const int b = blockIdx.z;
    const int h = blockIdx.y;
    const int qbase = blockIdx.x * 64 + wave * 16;
    const int kvh = h / (H_ / HKV_);
    const int causal = causal_p[0];

    const int colN = lane & 15;
    const int rowh = (lane >> 4) << 3;
    const int koff = (lane < 16) ? 0 : 8;
    const int kb2  = (lane >> 4) << 3;

    // ---- load Q fragments: 16x128 as 4 K-chunks of 32 ----
    bf16x16 aq[4];
    {
        const uint4* qrow = reinterpret_cast<const uint4*>(
            qb + ((size_t)(b * H_ + h) * T_ + qbase + colN) * HD_);
#pragma unroll
        for (int c = 0; c < 4; ++c) {
            uint4 v0 = qrow[c * 4 + (koff >> 3)];
            uint4 v1 = qrow[c * 4 + 2 + (koff >> 3)];
            unpack8(v0, aq[c], 0);
            unpack8(v1, aq[c], 8);
        }
    }

    f32x8 o[8];
    float m[8], l[8];
#pragma unroll
    for (int n = 0; n < 8; ++n)
#pragma unroll
        for (int r = 0; r < 8; ++r) o[n][r] = 0.0f;
#pragma unroll
    for (int r = 0; r < 8; ++r) { m[r] = -1.0e30f; l[r] = 0.0f; }

    const int kmax = causal ? (blockIdx.x * 64 + 64) : T_;
    const unsigned short* kt_g = kb + (size_t)(b * HKV_ + kvh) * T_ * HD_;
    const unsigned short* vt_g = vb + (size_t)(b * HKV_ + kvh) * T_ * HD_;
    const float scale = 0.08838834764831845f; // 1/sqrt(128)
    const unsigned ktBase = lds_off(&Kt[0]);

    for (int kt0 = 0; kt0 < kmax; kt0 += 32) {
        __syncthreads();
        // ---- K tile [32][128]: async global->LDS copy (byte-identical) ----
#pragma unroll
        for (int it = 0; it < 4; ++it) {
            int g = tid + it * 128;                  // 512 x 16B chunks
            unsigned dst = ktBase + (unsigned)g * 16u;
            unsigned long long src = (unsigned long long)(uintptr_t)(
                kt_g + (size_t)kt0 * HD_ + (size_t)g * 8);
            asm volatile("global_load_async_to_lds_b128 %0, %1, off"
                         :: "v"(dst), "v"(src) : "memory");
        }
        // ---- V tile transposed [hd][key] via regular loads ----
#pragma unroll
        for (int it = 0; it < 4; ++it) {
            int g   = tid + it * 128;
            int key = g >> 4;
            int hq  = g & 15;
            uint4 vv = reinterpret_cast<const uint4*>(
                vt_g + (size_t)(kt0 + key) * HD_)[hq];
            unsigned w[4] = { vv.x, vv.y, vv.z, vv.w };
            int hd0 = hq * 8;
#pragma unroll
            for (int j = 0; j < 4; ++j) {
                Vt[(hd0 + 2*j)     * 32 + key] = (unsigned short)(w[j] & 0xffffu);
                Vt[(hd0 + 2*j + 1) * 32 + key] = (unsigned short)(w[j] >> 16);
            }
        }
        asm volatile("s_wait_asynccnt 0x0" ::: "memory");
        __syncthreads();

        // ---- S = Q K^T: fragment pairs rotate through 2 buffers ----
        f32x8 s0, s1;
#pragma unroll
        for (int r = 0; r < 8; ++r) { s0[r] = 0.0f; s1[r] = 0.0f; }
        {
            const unsigned short* r0 = &Kt[colN * 128 + kb2];
            const unsigned short* r1 = &Kt[(colN + 16) * 128 + kb2];
            bf16x16 ba0, bb0, ba1, bb1;
            ldfrag(r0 + 0, ba0); ldfrag(r1 + 0, bb0);
            ldfrag(r0 + 32, ba1); ldfrag(r1 + 32, bb1);
            s0 = __builtin_amdgcn_wmma_f32_16x16x32_bf16(
                false, aq[0], false, ba0, (short)0, s0, false, false);
            s1 = __builtin_amdgcn_wmma_f32_16x16x32_bf16(
                false, aq[0], false, bb0, (short)0, s1, false, false);
            ldfrag(r0 + 64, ba0); ldfrag(r1 + 64, bb0);
            s0 = __builtin_amdgcn_wmma_f32_16x16x32_bf16(
                false, aq[1], false, ba1, (short)0, s0, false, false);
            s1 = __builtin_amdgcn_wmma_f32_16x16x32_bf16(
                false, aq[1], false, bb1, (short)0, s1, false, false);
            ldfrag(r0 + 96, ba1); ldfrag(r1 + 96, bb1);
            s0 = __builtin_amdgcn_wmma_f32_16x16x32_bf16(
                false, aq[2], false, ba0, (short)0, s0, false, false);
            s1 = __builtin_amdgcn_wmma_f32_16x16x32_bf16(
                false, aq[2], false, bb0, (short)0, s1, false, false);
            s0 = __builtin_amdgcn_wmma_f32_16x16x32_bf16(
                false, aq[3], false, ba1, (short)0, s0, false, false);
            s1 = __builtin_amdgcn_wmma_f32_16x16x32_bf16(
                false, aq[3], false, bb1, (short)0, s1, false, false);
        }

        // ---- online softmax (row = qbase + rowh + r; cols across 16 lanes) ----
        const int key0 = kt0 + colN;
        const int key1 = key0 + 16;
        const int am0 = amask[b * T_ + key0];
        const int am1 = amask[b * T_ + key1];
        float alpha[8];
#pragma unroll
        for (int r = 0; r < 8; ++r) {
            int trow = qbase + rowh + r;
            float v0 = s0[r] * scale;
            float v1 = s1[r] * scale;
            if (am0 == 0 || (causal && key0 > trow)) v0 = -1.0e30f;
            if (am1 == 0 || (causal && key1 > trow)) v1 = -1.0e30f;
            float mx = fmaxf(v0, v1);
#pragma unroll
            for (int off = 1; off < 16; off <<= 1)
                mx = fmaxf(mx, __shfl_xor(mx, off, 32));
            float nm = fmaxf(m[r], mx);
            float al = __expf(m[r] - nm);
            float p0 = __expf(v0 - nm);
            float p1 = __expf(v1 - nm);
            float ps = p0 + p1;
#pragma unroll
            for (int off = 1; off < 16; off <<= 1)
                ps += __shfl_xor(ps, off, 32);
            l[r] = l[r] * al + ps;
            m[r] = nm;
            alpha[r] = al;
            s0[r] = p0;
            s1[r] = p1;
        }
        // rescale accumulators
#pragma unroll
        for (int n = 0; n < 8; ++n)
#pragma unroll
            for (int r = 0; r < 8; ++r) o[n][r] *= alpha[r];

        // ---- P (C layout) -> LDS -> A layout bf16 fragment ----
        unsigned short* pw = &Ps[wave][0];
#pragma unroll
        for (int r = 0; r < 8; ++r) {
            pw[(rowh + r) * 32 + colN]      = f2us(s0[r]);
            pw[(rowh + r) * 32 + 16 + colN] = f2us(s1[r]);
        }
        asm volatile("s_wait_dscnt 0" ::: "memory");
        bf16x16 ap;
        {
            const unsigned short* pr = pw + colN * 32;
            uint4 v0 = *reinterpret_cast<const uint4*>(pr + koff);
            uint4 v1 = *reinterpret_cast<const uint4*>(pr + 16 + koff);
            unpack8(v0, ap, 0);
            unpack8(v1, ap, 8);
        }

        // ---- O += P V: fragments rotate through 2 buffers ----
        {
            const unsigned short* vbase = &Vt[colN * 32 + kb2];
            bf16x16 v0f, v1f;
            ldfrag(vbase + 0 * 512, v0f);
#pragma unroll
            for (int n = 0; n < 8; ++n) {
                if (n + 1 < 8) {
                    if (n & 1) ldfrag(vbase + (n + 1) * 512, v0f);
                    else       ldfrag(vbase + (n + 1) * 512, v1f);
                }
                o[n] = __builtin_amdgcn_wmma_f32_16x16x32_bf16(
                    false, ap, false, (n & 1) ? v1f : v0f, (short)0,
                    o[n], false, false);
            }
        }
    }

    // ---- epilogue: O / l, store f32 [b,t,h*hd] ----
    float invl[8];
#pragma unroll
    for (int r = 0; r < 8; ++r) invl[r] = (l[r] > 0.0f) ? (1.0f / l[r]) : 0.0f;
#pragma unroll
    for (int n = 0; n < 8; ++n)
#pragma unroll
        for (int r = 0; r < 8; ++r)
            attnf[(size_t)(b * T_ + qbase + rowh + r) * (H_ * HD_)
                  + h * HD_ + n * 16 + colN] = o[n][r] * invl[r];
}

// ---------------------------------------------------------------------------
extern "C" void kernel_launch(void* const* d_in, const int* in_sizes, int n_in,
                              void* d_out, int out_size, void* d_ws, size_t ws_size,
                              hipStream_t stream) {
    (void)in_sizes; (void)n_in; (void)out_size; (void)ws_size;
    const float* x      = (const float*)d_in[0];
    const int*   amask  = (const int*)d_in[1];
    const int*   causal = (const int*)d_in[2];
    const float* wq     = (const float*)d_in[3];
    const float* wk     = (const float*)d_in[4];
    const float* wv     = (const float*)d_in[5];
    const float* wo     = (const float*)d_in[6];
    float* out = (float*)d_out;

    char* ws = (char*)d_ws;
    // qf (32MB, aliased by attnf after rope), kf (8MB), vf (8MB),
    // qb bf16 (16MB), kb bf16 (4MB), vb bf16 (4MB)  => ~72MB
    float* qf = (float*)(ws + 0);
    float* attnf = qf;
    float* kf = (float*)(ws + (size_t)33554432);
    float* vf = (float*)(ws + (size_t)41943040);
    unsigned short* qbb = (unsigned short*)(ws + (size_t)50331648);
    unsigned short* kbb = (unsigned short*)(ws + (size_t)67108864);
    unsigned short* vbb = (unsigned short*)(ws + (size_t)71303168);

    const int M = B_ * T_;  // 4096
    dim3 blk(128);

    // QKV projections
    gemm_bf16_wmma<<<dim3((H_*HD_)/64,  M/64), blk, 0, stream>>>(x, wq, qf, M, H_*HD_,  D_);
    gemm_bf16_wmma<<<dim3((HKV_*HD_)/64, M/64), blk, 0, stream>>>(x, wk, kf, M, HKV_*HD_, D_);
    gemm_bf16_wmma<<<dim3((HKV_*HD_)/64, M/64), blk, 0, stream>>>(x, wv, vf, M, HKV_*HD_, D_);

    // RoPE + layout to [b,h,t,hd] bf16
    {
        int tq = B_ * H_ * T_ * 64;
        rope_to_bf16<<<(tq + 255) / 256, 256, 0, stream>>>(qf, qbb, H_, tq);
        int tk = B_ * HKV_ * T_ * 64;
        rope_to_bf16<<<(tk + 255) / 256, 256, 0, stream>>>(kf, kbb, HKV_, tk);
        int tv = B_ * HKV_ * T_ * HD_;
        v_to_bf16<<<(tv + 255) / 256, 256, 0, stream>>>(vf, vbb, tv);
    }

    // Flash attention (writes attnf, aliasing qf which is now consumed)
    gqa_flash_attn<<<dim3(T_/64, H_, B_), blk, 0, stream>>>(
        qbb, kbb, vbb, amask, causal, attnf);

    // Output projection
    gemm_bf16_wmma<<<dim3(D_/64, M/64), blk, 0, stream>>>(attnf, wo, out, M, D_, D_);
}
